// TextGenerationModel_13245679140996
// MI455X (gfx1250) — matrix operand order repaired and checked by reading, not compiled, not run on hardware
//
#include <hip/hip_runtime.h>
#include <hip/hip_bf16.h>

typedef __attribute__((ext_vector_type(16))) __bf16 v16bf;
typedef __attribute__((ext_vector_type(8)))  float  v8f;
typedef __attribute__((ext_vector_type(4)))  unsigned int u32x4;   // POD 16B vector
typedef __attribute__((ext_vector_type(4)))  int i32x4;

typedef const u32x4 __attribute__((address_space(1)))* gvec_ptr;   // global-AS 16B vector ptr
typedef const i32x4 __attribute__((address_space(1)))* givec_ptr;

#define H_  256
#define B_  512
#define S_  512
#define V_  128
#define NG_ 1024      // 4*H
#define KT_ 8         // 256 / 32
#define BM_ 32        // batch rows per workgroup
#define LDH_ 264      // padded LDS row stride (elements) to dodge bank conflicts

__device__ __forceinline__ unsigned short f2bf(float f) {
    unsigned u = __builtin_bit_cast(unsigned, f);
    u = (u + 0x7FFFu + ((u >> 16) & 1u)) >> 16;   // round-to-nearest-even
    return (unsigned short)u;
}

__device__ __forceinline__ float sig_(float x) { return 1.0f / (1.0f + __expf(-x)); }
// Branchless tanh, safe at +/-inf: t->inf => 1, t->0 => -1. One v_exp_f32.
__device__ __forceinline__ float tanh_(float x) {
    float t = __expf(2.0f * x);
    return 1.0f - 2.0f / (t + 1.0f);
}

union FragU { u32x4 u[2]; v16bf v; };

// Defeat LICM across the timestep loop: the base address is laundered through
// an empty asm each iteration, so 1.5MB of "loop-invariant" weight fragments
// can never be hoisted+spilled. Cast back to address_space(1) so loads stay
// global_load_b128 (SADDR form), not flat_load.
__device__ __forceinline__ unsigned long long launder_u64(unsigned long long a) {
    asm volatile("" : "+s"(a));
    return a;
}

// B-fragment at compile-time u32x4 index (folds into the 24-bit inst offset).
__device__ __forceinline__ v16bf load_b_at(gvec_ptr q, int idx) {
    FragU t; t.u[0] = q[idx]; t.u[1] = q[idx + 1];
    return t.v;
}

// A-fragment from per-lane LDS base pointer; kt folds into ds_load offsets.
__device__ __forceinline__ v16bf load_a_at(const __bf16* p, int kt) {
    FragU t;
    t.u[0] = *(const u32x4*)(p + kt * 32);
    t.u[1] = *(const u32x4*)(p + kt * 32 + 8);
    return t.v;
}

__device__ __forceinline__ v8f wmma_bf16(v16bf a, v16bf b, v8f c) {
    return __builtin_amdgcn_wmma_f32_16x16x32_bf16(false, a, false, b, (short)0, c, false, false);
}

// ---------------------------------------------------------------------------
// Pack: fp32 weights [N][K=256] -> bf16 in WMMA B-fragment order; fold biases.
// packed[((nt*8+kt)*32 + l)*16 + j] = w[nt*16 + (l&15)][kt*32 + (l>=16)*16 + j]
// ---------------------------------------------------------------------------
__global__ void pack_weights(const float* __restrict__ w_hh0, const float* __restrict__ w_ih1,
                             const float* __restrict__ w_hh1, const float* __restrict__ W_out,
                             const float* __restrict__ b_ih0, const float* __restrict__ b_hh0,
                             const float* __restrict__ b_ih1, const float* __restrict__ b_hh1,
                             unsigned short* __restrict__ pw_hh0, unsigned short* __restrict__ pw_ih1,
                             unsigned short* __restrict__ pw_hh1, unsigned short* __restrict__ pW_out,
                             float* __restrict__ bias0, float* __restrict__ bias1) {
    const int PG = NG_ * H_;   // 262144 packed elements per gate-weight matrix
    const int PO = V_ * H_;    // 32768
    int idx = blockIdx.x * blockDim.x + threadIdx.x;
    if (idx < 3 * PG) {
        const float* src; unsigned short* dst; int e = idx;
        if (e < PG)          { src = w_hh0; dst = pw_hh0; }
        else if (e < 2 * PG) { src = w_ih1; dst = pw_ih1; e -= PG; }
        else                 { src = w_hh1; dst = pw_hh1; e -= 2 * PG; }
        int j  = e & 15;
        int l  = (e >> 4) & 31;
        int kt = (e >> 9) & 7;
        int nt = e >> 12;
        int n = nt * 16 + (l & 15);
        int k = kt * 32 + ((l >> 4) << 4) + j;
        dst[e] = f2bf(src[n * H_ + k]);
    } else if (idx < 3 * PG + PO) {
        int e = idx - 3 * PG;
        int j  = e & 15;
        int l  = (e >> 4) & 31;
        int kt = (e >> 9) & 7;
        int nt = e >> 12;
        int n = nt * 16 + (l & 15);
        int k = kt * 32 + ((l >> 4) << 4) + j;
        pW_out[e] = f2bf(W_out[n * H_ + k]);
    } else if (idx < 3 * PG + PO + NG_) {
        int e = idx - 3 * PG - PO;
        bias0[e] = b_ih0[e] + b_hh0[e];
    } else if (idx < 3 * PG + PO + 2 * NG_) {
        int e = idx - 3 * PG - PO - NG_;
        bias1[e] = b_ih1[e] + b_hh1[e];
    }
}

// ---------------------------------------------------------------------------
// Persistent 2-layer LSTM + fused vocab projection.
// Grid: 16 WGs (batch slices of 32). Block: 256 threads = 8 waves.
// Wave w owns hidden slice [32w, 32w+32): gate tiles g*16 + 2w + ht, so
// i/f/g/o for a given hidden unit all live in the same wave's accumulators.
// ---------------------------------------------------------------------------
__global__ __launch_bounds__(256, 1) void lstm_main(
    const int* __restrict__ x, const float* __restrict__ w_ih0,
    const unsigned short* pw_hh0_, const unsigned short* pw_ih1_,
    const unsigned short* pw_hh1_, const unsigned short* pW_out_,
    const float* __restrict__ bias0, const float* __restrict__ bias1,
    const float* __restrict__ b_out, float* __restrict__ out) {
    __shared__ __bf16 sh0[BM_ * LDH_];   // h0 (bf16, A-operand layout source)
    __shared__ __bf16 sh1[BM_ * LDH_];   // h1

    const int tid   = threadIdx.x;
    const int lane  = tid & 31;
    const int wave  = tid >> 5;
    const int col   = lane & 15;          // N column within 16x16 C tile
    const int rbase = (lane >> 4) << 3;   // rows 0-7 or 8-15 of C tile
    const int hb    = wave * 32;          // hidden-slice base for gates
    const int vb    = wave * 16;          // vocab-slice base for output
    const int batch0 = blockIdx.x * BM_;

    // Loop-invariant per-lane u32x4 indices into the packed weight arrays:
    // gate matrices: fold wave (16 KB) + lane (32 B) into the voffset;
    // W_out: wave stride is 8 KB. x: per-lane row group (rbase) in i32x4 units.
    const int goff = wave * 1024 + lane * 2;   // u32x4 units
    const int ooff = wave * 512  + lane * 2;
    const int xoff = rbase >> 2;               // i32x4 units (0 or 2)

    // Loop-invariant per-lane LDS A-fragment base pointers (mt = 0 / 1).
    const int arow = (lane & 15) * LDH_ + ((lane >> 4) << 4);
    const __bf16* pa0_m0 = sh0 + arow;
    const __bf16* pa0_m1 = sh0 + 16 * LDH_ + arow;
    const __bf16* pa1_m0 = sh1 + arow;
    const __bf16* pa1_m1 = sh1 + 16 * LDH_ + arow;

    // Per-lane scalar constants (each lane's gate column is fixed).
    float b0v[4][2], b1v[4][2], wi0v[4][2];
#pragma unroll
    for (int g = 0; g < 4; ++g)
#pragma unroll
        for (int ht = 0; ht < 2; ++ht) {
            int n = g * H_ + hb + ht * 16 + col;
            b0v[g][ht]  = bias0[n];
            b1v[g][ht]  = bias1[n];
            wi0v[g][ht] = w_ih0[n];       // w_ih0 is [4H, 1]
        }
    float boutv = b_out[vb + col];

    for (int i = tid; i < BM_ * LDH_; i += 256) { sh0[i] = (__bf16)0.0f; sh1[i] = (__bf16)0.0f; }
    __syncthreads();

    // Cell states in accumulator-shaped registers.
    v8f c0[2][2], c1[2][2];
#pragma unroll
    for (int mt = 0; mt < 2; ++mt)
#pragma unroll
        for (int ht = 0; ht < 2; ++ht)
#pragma unroll
            for (int r = 0; r < 8; ++r) { c0[mt][ht][r] = 0.0f; c1[mt][ht][r] = 0.0f; }

#pragma clang loop unroll(disable)
    for (int t = 0; t < S_; ++t) {
        // Fresh laundered bases each iteration (SGPR, global address space).
        gvec_ptr qh0 = (gvec_ptr)launder_u64((unsigned long long)pw_hh0_) + goff;
        gvec_ptr qi1 = (gvec_ptr)launder_u64((unsigned long long)pw_ih1_) + goff;
        gvec_ptr qh1 = (gvec_ptr)launder_u64((unsigned long long)pw_hh1_) + goff;
        gvec_ptr qo  = (gvec_ptr)launder_u64((unsigned long long)pW_out_) + ooff;

        // Per-lane token loads: rows mt*16 + rbase + 0..7 are 8 consecutive
        // ints -> two b128 loads per mt, SADDR form, broadcast-friendly.
        givec_ptr xq = (givec_ptr)(unsigned long long)(x + t * B_ + batch0) + xoff;
        float xr[2][8];
#pragma unroll
        for (int mt = 0; mt < 2; ++mt) {
            i32x4 q0 = xq[mt * 4 + 0];
            i32x4 q1 = xq[mt * 4 + 1];
#pragma unroll
            for (int r = 0; r < 4; ++r) { xr[mt][r] = (float)q0[r]; xr[mt][4 + r] = (float)q1[r]; }
        }

        // ===== layer 0: gates = bias0 + w_ih0*x + h0 @ w_hh0^T =====
        v8f acc[2][4][2];
#pragma unroll
        for (int mt = 0; mt < 2; ++mt)
#pragma unroll
            for (int g = 0; g < 4; ++g)
#pragma unroll
                for (int ht = 0; ht < 2; ++ht)
#pragma unroll
                    for (int r = 0; r < 8; ++r)
                        acc[mt][g][ht][r] = b0v[g][ht] + wi0v[g][ht] * xr[mt][r];

#pragma unroll
        for (int kt = 0; kt < KT_; ++kt) {
            v16bf a0 = load_a_at(pa0_m0, kt);
            v16bf a1 = load_a_at(pa0_m1, kt);
#pragma unroll
            for (int g = 0; g < 4; ++g)
#pragma unroll
                for (int ht = 0; ht < 2; ++ht) {
                    v16bf b = load_b_at(qh0, g * 8192 + ht * 512 + kt * 64);
                    acc[0][g][ht] = wmma_bf16(a0, b, acc[0][g][ht]);
                    acc[1][g][ht] = wmma_bf16(a1, b, acc[1][g][ht]);
                }
        }
        __syncthreads();   // everyone finished reading old sh0

#pragma unroll
        for (int mt = 0; mt < 2; ++mt)
#pragma unroll
            for (int ht = 0; ht < 2; ++ht)
#pragma unroll
                for (int r = 0; r < 8; ++r) {
                    float iv = sig_(acc[mt][0][ht][r]);
                    float fv = sig_(acc[mt][1][ht][r]);
                    float gv = tanh_(acc[mt][2][ht][r]);
                    float ov = sig_(acc[mt][3][ht][r]);
                    float cn = fv * c0[mt][ht][r] + iv * gv;
                    c0[mt][ht][r] = cn;
                    float hn = ov * tanh_(cn);
                    sh0[(mt * 16 + rbase + r) * LDH_ + hb + ht * 16 + col] = (__bf16)hn;
                }
        __syncthreads();   // new h0 visible

        // ===== layer 1: gates = bias1 + h0 @ w_ih1^T + h1 @ w_hh1^T =====
#pragma unroll
        for (int mt = 0; mt < 2; ++mt)
#pragma unroll
            for (int g = 0; g < 4; ++g)
#pragma unroll
                for (int ht = 0; ht < 2; ++ht)
#pragma unroll
                    for (int r = 0; r < 8; ++r) acc[mt][g][ht][r] = b1v[g][ht];

#pragma unroll
        for (int kt = 0; kt < KT_; ++kt) {
            v16bf a0 = load_a_at(pa0_m0, kt);
            v16bf a1 = load_a_at(pa0_m1, kt);
#pragma unroll
            for (int g = 0; g < 4; ++g)
#pragma unroll
                for (int ht = 0; ht < 2; ++ht) {
                    v16bf b = load_b_at(qi1, g * 8192 + ht * 512 + kt * 64);
                    acc[0][g][ht] = wmma_bf16(a0, b, acc[0][g][ht]);
                    acc[1][g][ht] = wmma_bf16(a1, b, acc[1][g][ht]);
                }
        }
#pragma unroll
        for (int kt = 0; kt < KT_; ++kt) {
            v16bf a0 = load_a_at(pa1_m0, kt);
            v16bf a1 = load_a_at(pa1_m1, kt);
#pragma unroll
            for (int g = 0; g < 4; ++g)
#pragma unroll
                for (int ht = 0; ht < 2; ++ht) {
                    v16bf b = load_b_at(qh1, g * 8192 + ht * 512 + kt * 64);
                    acc[0][g][ht] = wmma_bf16(a0, b, acc[0][g][ht]);
                    acc[1][g][ht] = wmma_bf16(a1, b, acc[1][g][ht]);
                }
        }
        __syncthreads();   // everyone finished reading old sh1

#pragma unroll
        for (int mt = 0; mt < 2; ++mt)
#pragma unroll
            for (int ht = 0; ht < 2; ++ht)
#pragma unroll
                for (int r = 0; r < 8; ++r) {
                    float iv = sig_(acc[mt][0][ht][r]);
                    float fv = sig_(acc[mt][1][ht][r]);
                    float gv = tanh_(acc[mt][2][ht][r]);
                    float ov = sig_(acc[mt][3][ht][r]);
                    float cn = fv * c1[mt][ht][r] + iv * gv;
                    c1[mt][ht][r] = cn;
                    float hn = ov * tanh_(cn);
                    sh1[(mt * 16 + rbase + r) * LDH_ + hb + ht * 16 + col] = (__bf16)hn;
                }
        __syncthreads();   // new h1 visible

        // ===== fused output projection: logits = h1 @ W_out^T + b_out =====
        v8f oacc[2];
#pragma unroll
        for (int mt = 0; mt < 2; ++mt)
#pragma unroll
            for (int r = 0; r < 8; ++r) oacc[mt][r] = boutv;
#pragma unroll
        for (int kt = 0; kt < KT_; ++kt) {
            v16bf a0 = load_a_at(pa1_m0, kt);
            v16bf a1 = load_a_at(pa1_m1, kt);
            v16bf b  = load_b_at(qo, kt * 64);
            oacc[0] = wmma_bf16(a0, b, oacc[0]);
            oacc[1] = wmma_bf16(a1, b, oacc[1]);
        }
#pragma unroll
        for (int mt = 0; mt < 2; ++mt)
#pragma unroll
            for (int r = 0; r < 8; ++r) {
                int row = mt * 16 + rbase + r;
                out[(((size_t)t) * B_ + batch0 + row) * V_ + vb + col] = oacc[mt][r];
            }
        // next iteration's barriers protect sh0/sh1 reuse
    }
}

extern "C" void kernel_launch(void* const* d_in, const int* in_sizes, int n_in,
                              void* d_out, int out_size, void* d_ws, size_t ws_size,
                              hipStream_t stream) {
    const int*   x     = (const int*)d_in[0];
    const float* w_ih0 = (const float*)d_in[1];
    const float* w_hh0 = (const float*)d_in[2];
    const float* b_ih0 = (const float*)d_in[3];
    const float* b_hh0 = (const float*)d_in[4];
    const float* w_ih1 = (const float*)d_in[5];
    const float* w_hh1 = (const float*)d_in[6];
    const float* b_ih1 = (const float*)d_in[7];
    const float* b_hh1 = (const float*)d_in[8];
    const float* W_out = (const float*)d_in[9];
    const float* b_out = (const float*)d_in[10];
    float* out = (float*)d_out;

    char* ws = (char*)d_ws;
    unsigned short* pw_hh0 = (unsigned short*)(ws);             // 512 KB
    unsigned short* pw_ih1 = (unsigned short*)(ws + 524288);    // 512 KB
    unsigned short* pw_hh1 = (unsigned short*)(ws + 1048576);   // 512 KB
    unsigned short* pW_out = (unsigned short*)(ws + 1572864);   //  64 KB
    float* bias0 = (float*)(ws + 1638400);                      //   4 KB
    float* bias1 = (float*)(ws + 1642496);                      //   4 KB

    const int total = 3 * (NG_ * H_) + (V_ * H_) + 2 * NG_;
    pack_weights<<<(total + 255) / 256, 256, 0, stream>>>(
        w_hh0, w_ih1, w_hh1, W_out, b_ih0, b_hh0, b_ih1, b_hh1,
        pw_hh0, pw_ih1, pw_hh1, pW_out, bias0, bias1);

    lstm_main<<<B_ / BM_, 256, 0, stream>>>(
        x, w_ih0, pw_hh0, pw_ih1, pw_hh1, pW_out, bias0, bias1, b_out, out);
}
